// LSTMs4WithAttention_91164975825191
// MI455X (gfx1250) — matrix operand order repaired and checked by reading, not compiled, not run on hardware
//
#include <hip/hip_runtime.h>
#include <hip/hip_bf16.h>
#include <stdint.h>

typedef __bf16 bf16_t;
typedef __bf16 v16bf __attribute__((ext_vector_type(16)));
typedef __bf16 v8bf  __attribute__((ext_vector_type(8)));
typedef float  v8f   __attribute__((ext_vector_type(8)));

#define BB   128
#define TT   512
#define XDIM 266
#define COMB 192
#define NCLS 250

__device__ __forceinline__ float fsigmoid(float x) { return 1.f / (1.f + __expf(-x)); }
__device__ __forceinline__ float ftanh(float x)    { float e = __expf(2.f * x); return (e - 1.f) / (e + 1.f); }

// ---------------------------------------------------------------------------
// gfx1250 async global->LDS copy (16B per lane), ASYNCcnt-tracked.
// Builtin signature (from clang diagnostic): V4i addrspace(1)*, V4i addrspace(3)*,
// imm offset, imm cpol — V4i is a generic vector_size(16) int vector.
// Falls back to a synchronous copy with identical buffering if unavailable.
// ---------------------------------------------------------------------------
typedef int v4i_t __attribute__((vector_size(16)));
typedef __attribute__((address_space(1))) v4i_t as1_v4i;
typedef __attribute__((address_space(3))) v4i_t as3_v4i;

__device__ __forceinline__ void async_copy16(const float* g, float* l)
{
#if __has_builtin(__builtin_amdgcn_global_load_async_to_lds_b128)
    __builtin_amdgcn_global_load_async_to_lds_b128(
        (as1_v4i*)(uintptr_t)g,
        (as3_v4i*)(unsigned)(uintptr_t)l,
        0, 0);
#else
    *(float4*)l = *(const float4*)g;
#endif
}

__device__ __forceinline__ void async_wait_all()
{
#if __has_builtin(__builtin_amdgcn_s_wait_asynccnt)
    __builtin_amdgcn_s_wait_asynccnt(0);
#else
    asm volatile("s_wait_asynccnt 0x0" ::: "memory");
#endif
}

// ---------------------------------------------------------------------------
// Kernel 1: feature preparation (normalize point groups, write padded bf16)
// ---------------------------------------------------------------------------
__device__ __forceinline__ void norm_store(const float* __restrict__ row,
                                           int start, int step, int npts, int refp,
                                           bf16_t* __restrict__ dst, int kpad)
{
    float refx = row[2 * refp], refy = row[2 * refp + 1];
    float mnx = 3.4e38f, mxx = -3.4e38f, mny = 3.4e38f, mxy = -3.4e38f;
    for (int p = 0; p < npts; ++p) {
        int q = start + p * step;
        float vx = row[2 * q] - refx;
        float vy = row[2 * q + 1] - refy;
        mnx = fminf(mnx, vx); mxx = fmaxf(mxx, vx);
        mny = fminf(mny, vy); mxy = fmaxf(mxy, vy);
    }
    float s = fmaxf(mxx - mnx, mxy - mny);
    if (s == 0.f) s = 1.f;
    float inv = 1.f / s;
    int k = 0;
    for (int p = 0; p < npts; ++p) {
        int q = start + p * step;
        dst[k++] = (bf16_t)((row[2 * q] - refx) * inv);
        dst[k++] = (bf16_t)((row[2 * q + 1] - refy) * inv);
    }
    for (; k < kpad; ++k) dst[k] = (bf16_t)0.f;
}

__global__ void __launch_bounds__(256)
prep_kernel(const float* __restrict__ x,
            bf16_t* __restrict__ fRH, bf16_t* __restrict__ fRA,
            bf16_t* __restrict__ fLH, bf16_t* __restrict__ fLA)
{
    int id = blockIdx.x * 256 + threadIdx.x;
    if (id >= BB * TT) return;
    const float* row = x + (size_t)id * XDIM;
    norm_store(row, 112, 1, 21, 10, fRH + (size_t)id * 64, 64);  // right hand
    norm_store(row,   6, 2,  3,  0, fRA + (size_t)id * 32, 32);  // right arm (6,8,10)
    norm_store(row,  91, 1, 21,  9, fLH + (size_t)id * 64, 64);  // left hand
    norm_store(row,   5, 2,  3,  0, fLA + (size_t)id * 32, 32);  // left arm (5,7,9)
}

// ---------------------------------------------------------------------------
// Kernel 2: input projection GEMM  pre = feat @ wih^T + bih   (bf16 WMMA)
// grid.x tiles M = B*T by 128 rows (8 waves x 16 rows); grid.y picks LSTM.
// ---------------------------------------------------------------------------
template <int KP, int G, int F>
__global__ void __launch_bounds__(256)
inproj_kernel(const bf16_t* __restrict__ featA, const float* __restrict__ wihA,
              const float* __restrict__ bihA, float* __restrict__ preA,
              const bf16_t* __restrict__ featB2, const float* __restrict__ wihB2,
              const float* __restrict__ bihB2, float* __restrict__ preB2)
{
    const bf16_t* feat = blockIdx.y ? featB2 : featA;
    const float*  wih  = blockIdx.y ? wihB2  : wihA;
    const float*  bih  = blockIdx.y ? bihB2  : bihA;
    float*        pre  = blockIdx.y ? preB2  : preA;

    __shared__ __align__(16) bf16_t lds_w[G * KP];   // [n][k], K padded, bf16
    int tid = threadIdx.x;
    for (int idx = tid; idx < G * KP; idx += 256) {
        int n = idx / KP, k = idx % KP;
        lds_w[idx] = (bf16_t)((k < F) ? wih[n * F + k] : 0.f);
    }
    __syncthreads();

    int wave = tid >> 5, lane = tid & 31;
    int lr = lane & 15;
    int hi = (lane >> 4) & 1;
    int m0 = blockIdx.x * 128 + wave * 16;

    constexpr int NK = KP / 32;
    v16bf afrag[NK];
#pragma unroll
    for (int kk = 0; kk < NK; ++kk) {
        const bf16_t* arow = feat + (size_t)(m0 + lr) * KP + kk * 32 + hi * 8;
        v8bf lo = *(const v8bf*)(arow);
        v8bf hv = *(const v8bf*)(arow + 16);
#pragma unroll
        for (int e = 0; e < 8; ++e) { afrag[kk][e] = lo[e]; afrag[kk][e + 8] = hv[e]; }
    }

#pragma unroll
    for (int nt = 0; nt < G / 16; ++nt) {
        int n = nt * 16 + lr;
        float bb = bih[n];
        v8f acc;
#pragma unroll
        for (int r = 0; r < 8; ++r) acc[r] = bb;
#pragma unroll
        for (int kk = 0; kk < NK; ++kk) {
            const bf16_t* brow = lds_w + n * KP + kk * 32 + hi * 8;
            v8bf b0 = *(const v8bf*)(brow);
            v8bf b1 = *(const v8bf*)(brow + 16);
            v16bf bf;
#pragma unroll
            for (int e = 0; e < 8; ++e) { bf[e] = b0[e]; bf[e + 8] = b1[e]; }
            acc = __builtin_amdgcn_wmma_f32_16x16x32_bf16(false, afrag[kk], false, bf,
                                                          (short)0, acc, false, false);
        }
#pragma unroll
        for (int r = 0; r < 8; ++r) {
            int m = m0 + r + hi * 8;
            pre[(size_t)m * G + n] = acc[r];
        }
    }
}

// ---------------------------------------------------------------------------
// Kernel 3: recurrent LSTM. One block = 16 batch rows; whh B-fragments live in
// VGPRs for the whole T loop; h in LDS (bf16); cell state in registers.
// pre[t] gate inputs are double-buffered in LDS via async global->LDS copies
// issued one timestep ahead (ASYNCcnt), hiding HBM latency in the recurrence.
// grid.x = B/16, grid.y picks which of the two same-H LSTMs.
// ---------------------------------------------------------------------------
template <int H>
__global__ void __launch_bounds__(128)
lstm_kernel(const float* __restrict__ preA, const float* __restrict__ whhA,
            const float* __restrict__ bhhA, int coA,
            const float* __restrict__ preB2, const float* __restrict__ whhB2,
            const float* __restrict__ bhhB2, int coB,
            float* __restrict__ combined)
{
    constexpr int G   = 4 * H;
    constexpr int NK  = H / 32;          // K steps per GEMM
    constexpr int NT  = G / 16;          // N tiles total
    constexpr int NTW = NT / 4;          // N tiles per wave (4 waves)
    constexpr int CPT = (16 * H) / 128;  // cells per thread
    constexpr int CH  = (16 * G) / (4 * 128); // 16B chunks per thread per tile

    const float* pre = blockIdx.y ? preB2 : preA;
    const float* whh = blockIdx.y ? whhB2 : whhA;
    const float* bhh = blockIdx.y ? bhhB2 : bhhA;
    int co = blockIdx.y ? coB : coA;

    __shared__ __align__(16) bf16_t lds_w[G * H];    // whh as [n][k] bf16
    __shared__ __align__(16) bf16_t lds_h[16 * H];   // hidden state, bf16
    __shared__ __align__(16) float  lds_g[16 * G];   // gate pre-activations
    __shared__ __align__(16) float  lds_p0[16 * G];  // pre[t] double buffer 0
    __shared__ __align__(16) float  lds_p1[16 * G];  // pre[t] double buffer 1

    int tid = threadIdx.x;
    for (int idx = tid; idx < G * H; idx += 128) lds_w[idx] = (bf16_t)whh[idx];
    for (int idx = tid; idx < 16 * H; idx += 128) lds_h[idx] = (bf16_t)0.f;

    int wave = tid >> 5, lane = tid & 31;
    int lr = lane & 15;
    int hi = (lane >> 4) & 1;
    int m0 = blockIdx.x * 16;

    // Per-thread base pointers for the async pre-tile copies.
    const float* gbase[CH];
    float* l0[CH];
    float* l1[CH];
#pragma unroll
    for (int k2 = 0; k2 < CH; ++k2) {
        int idx = tid + k2 * 128;
        int m = idx / (G / 4);
        int c4 = (idx % (G / 4)) * 4;
        gbase[k2] = pre + (size_t)(m0 + m) * TT * G + c4;
        l0[k2] = lds_p0 + m * G + c4;
        l1[k2] = lds_p1 + m * G + c4;
    }
    // Prologue: start the copy of pre[0] into buffer 0.
#pragma unroll
    for (int k2 = 0; k2 < CH; ++k2) async_copy16(gbase[k2], l0[k2]);
    __syncthreads();

    v16bf bfrag[NTW * NK];  // whh^T fragments, resident for the whole T loop
    float bias[NTW];
#pragma unroll
    for (int j = 0; j < NTW; ++j) {
        int n = (wave * NTW + j) * 16 + lr;
        bias[j] = bhh[n];
#pragma unroll
        for (int kk = 0; kk < NK; ++kk) {
            const bf16_t* brow = lds_w + n * H + kk * 32 + hi * 8;
            v8bf b0 = *(const v8bf*)(brow);
            v8bf b1 = *(const v8bf*)(brow + 16);
#pragma unroll
            for (int e = 0; e < 8; ++e) {
                bfrag[j * NK + kk][e] = b0[e];
                bfrag[j * NK + kk][e + 8] = b1[e];
            }
        }
    }

    float cst[CPT];
#pragma unroll
    for (int j = 0; j < CPT; ++j) cst[j] = 0.f;

    for (int t = 0; t < TT; ++t) {
        async_wait_all();     // this wave's copy of pre[t] has landed in LDS
        __syncthreads();      // all waves' chunks landed; prev-step lds_h visible

        // Kick off the async copy for pre[t+1] into the other buffer.
        if (t + 1 < TT) {
            if ((t + 1) & 1) {
#pragma unroll
                for (int k2 = 0; k2 < CH; ++k2)
                    async_copy16(gbase[k2] + (size_t)(t + 1) * G, l1[k2]);
            } else {
#pragma unroll
                for (int k2 = 0; k2 < CH; ++k2)
                    async_copy16(gbase[k2] + (size_t)(t + 1) * G, l0[k2]);
            }
        }
        const float* lds_pre = (t & 1) ? lds_p1 : lds_p0;

        // A fragments from current hidden state in LDS.
        v16bf afrag[NK];
#pragma unroll
        for (int kk = 0; kk < NK; ++kk) {
            const bf16_t* arow = lds_h + lr * H + kk * 32 + hi * 8;
            v8bf lo = *(const v8bf*)(arow);
            v8bf hv = *(const v8bf*)(arow + 16);
#pragma unroll
            for (int e = 0; e < 8; ++e) { afrag[kk][e] = lo[e]; afrag[kk][e + 8] = hv[e]; }
        }
#pragma unroll
        for (int j = 0; j < NTW; ++j) {
            int n = (wave * NTW + j) * 16 + lr;
            v8f acc;
#pragma unroll
            for (int r = 0; r < 8; ++r) {
                int m = r + hi * 8;
                acc[r] = lds_pre[m * G + n] + bias[j];
            }
#pragma unroll
            for (int kk = 0; kk < NK; ++kk)
                acc = __builtin_amdgcn_wmma_f32_16x16x32_bf16(false, afrag[kk], false,
                                                              bfrag[j * NK + kk],
                                                              (short)0, acc, false, false);
#pragma unroll
            for (int r = 0; r < 8; ++r) {
                int m = r + hi * 8;
                lds_g[m * G + n] = acc[r];
            }
        }
        __syncthreads();      // gates visible to every thread
        // Elementwise gate math; cell state stays in registers.
#pragma unroll
        for (int j = 0; j < CPT; ++j) {
            int cidx = tid * CPT + j;
            int m = cidx / H, hh = cidx % H;
            float gi = fsigmoid(lds_g[m * G + hh]);
            float gf = fsigmoid(lds_g[m * G + H + hh]);
            float gg = ftanh(lds_g[m * G + 2 * H + hh]);
            float go = fsigmoid(lds_g[m * G + 3 * H + hh]);
            float c = gf * cst[j] + gi * gg;
            cst[j] = c;
            float hval = go * ftanh(c);
            lds_h[m * H + hh] = (bf16_t)hval;
            combined[((size_t)(m0 + m) * TT + t) * COMB + co + hh] = hval;
        }
        // Next iteration's top barrier orders these lds_h writes.
    }
}

// ---------------------------------------------------------------------------
// Kernel 4: attention softmax over T + context + FC. One block per batch.
// ---------------------------------------------------------------------------
__global__ void __launch_bounds__(256)
attn_fc_kernel(const float* __restrict__ combined, const float* __restrict__ att_w,
               const float* __restrict__ fc_w, const float* __restrict__ fc_b,
               float* __restrict__ out)
{
    int b = blockIdx.x, tid = threadIdx.x;
    __shared__ float s_sc[TT];
    __shared__ float s_red[256];
    __shared__ float s_aw[COMB];
    __shared__ float s_ctx[COMB];
    if (tid < COMB) s_aw[tid] = att_w[tid];
    __syncthreads();

    const float* cb = combined + (size_t)b * TT * COMB;
    float lmax = -3.4e38f;
    for (int t = tid; t < TT; t += 256) {
        const float* rowp = cb + (size_t)t * COMB;
        float s = 0.f;
        for (int d = 0; d < COMB; ++d) s += rowp[d] * s_aw[d];
        s_sc[t] = s;
        lmax = fmaxf(lmax, s);
    }
    s_red[tid] = lmax; __syncthreads();
    for (int off = 128; off > 0; off >>= 1) {
        if (tid < off) s_red[tid] = fmaxf(s_red[tid], s_red[tid + off]);
        __syncthreads();
    }
    float mx = s_red[0];
    float lsum = 0.f;
    for (int t = tid; t < TT; t += 256) {
        float e = __expf(s_sc[t] - mx);
        s_sc[t] = e;
        lsum += e;
    }
    __syncthreads();
    s_red[tid] = lsum; __syncthreads();
    for (int off = 128; off > 0; off >>= 1) {
        if (tid < off) s_red[tid] += s_red[tid + off];
        __syncthreads();
    }
    float inv = 1.f / s_red[0];
    for (int t = tid; t < TT; t += 256) {
        float w = s_sc[t] * inv;
        s_sc[t] = w;
        out[(size_t)BB * NCLS + (size_t)b * TT + t] = w;  // weights output
    }
    __syncthreads();
    if (tid < COMB) {
        float acc = 0.f;
        for (int t = 0; t < TT; ++t) acc += s_sc[t] * cb[(size_t)t * COMB + tid];
        s_ctx[tid] = acc;
    }
    __syncthreads();
    if (tid < NCLS) {
        const float* wr = fc_w + (size_t)tid * COMB;
        float acc = fc_b[tid];
        for (int d = 0; d < COMB; ++d) acc += wr[d] * s_ctx[d];
        out[(size_t)b * NCLS + tid] = acc;  // logits output
    }
}

// ---------------------------------------------------------------------------
extern "C" void kernel_launch(void* const* d_in, const int* in_sizes, int n_in,
                              void* d_out, int out_size, void* d_ws, size_t ws_size,
                              hipStream_t stream)
{
    (void)in_sizes; (void)n_in; (void)out_size; (void)ws_size;
    const float* x      = (const float*)d_in[0];
    const float* wih_rh = (const float*)d_in[1];
    const float* whh_rh = (const float*)d_in[2];
    const float* bih_rh = (const float*)d_in[3];
    const float* bhh_rh = (const float*)d_in[4];
    const float* wih_ra = (const float*)d_in[5];
    const float* whh_ra = (const float*)d_in[6];
    const float* bih_ra = (const float*)d_in[7];
    const float* bhh_ra = (const float*)d_in[8];
    const float* wih_lh = (const float*)d_in[9];
    const float* whh_lh = (const float*)d_in[10];
    const float* bih_lh = (const float*)d_in[11];
    const float* bhh_lh = (const float*)d_in[12];
    const float* wih_la = (const float*)d_in[13];
    const float* whh_la = (const float*)d_in[14];
    const float* bih_la = (const float*)d_in[15];
    const float* bhh_la = (const float*)d_in[16];
    const float* att_w  = (const float*)d_in[17];
    const float* fc_w   = (const float*)d_in[18];
    const float* fc_b   = (const float*)d_in[19];
    float* out = (float*)d_out;

    const size_t M = (size_t)BB * TT;
    char* ws = (char*)d_ws;
    size_t off = 0;
    bf16_t* fRH = (bf16_t*)(ws + off); off += M * 64 * sizeof(bf16_t);
    bf16_t* fRA = (bf16_t*)(ws + off); off += M * 32 * sizeof(bf16_t);
    bf16_t* fLH = (bf16_t*)(ws + off); off += M * 64 * sizeof(bf16_t);
    bf16_t* fLA = (bf16_t*)(ws + off); off += M * 32 * sizeof(bf16_t);
    float* preRH = (float*)(ws + off); off += M * 256 * sizeof(float);
    float* preRA = (float*)(ws + off); off += M * 128 * sizeof(float);
    float* preLH = (float*)(ws + off); off += M * 256 * sizeof(float);
    float* preLA = (float*)(ws + off); off += M * 128 * sizeof(float);
    float* combined = (float*)(ws + off); off += M * COMB * sizeof(float);

    // 1) feature prep
    prep_kernel<<<(BB * TT) / 256, 256, 0, stream>>>(x, fRH, fRA, fLH, fLA);

    // 2) input projections (rh+lh share a launch; ra+la share a launch)
    dim3 g2((unsigned)(M / 128), 2);
    inproj_kernel<64, 256, 42><<<g2, 256, 0, stream>>>(fRH, wih_rh, bih_rh, preRH,
                                                       fLH, wih_lh, bih_lh, preLH);
    inproj_kernel<32, 128, 6><<<g2, 256, 0, stream>>>(fRA, wih_ra, bih_ra, preRA,
                                                      fLA, wih_la, bih_la, preLA);

    // 3) recurrent LSTMs. combined layout: [rh(0..63) | ra(64..95) | lh(96..159) | la(160..191)]
    dim3 g3(BB / 16, 2);
    lstm_kernel<64><<<g3, 128, 0, stream>>>(preRH, whh_rh, bhh_rh, 0,
                                            preLH, whh_lh, bhh_lh, 96, combined);
    lstm_kernel<32><<<g3, 128, 0, stream>>>(preRA, whh_ra, bhh_ra, 64,
                                            preLA, whh_la, bhh_la, 160, combined);

    // 4) attention + FC
    attn_fc_kernel<<<BB, 256, 0, stream>>>(combined, att_w, fc_w, fc_b, out);
}
